// CrossAttention_47485158425033
// MI455X (gfx1250) — matrix operand order repaired and checked by reading, not compiled, not run on hardware
//
#include <hip/hip_runtime.h>

typedef __attribute__((ext_vector_type(16))) __bf16 v16bf;
typedef __attribute__((ext_vector_type(8)))  float  v8f;

union FragU { v16bf v; uint4 q[2]; };

static __device__ __forceinline__ unsigned short f2bf(float f) {
  union { float f; unsigned int u; } x; x.f = f;
  unsigned int u = x.u + 0x7FFFu + ((x.u >> 16) & 1u);
  return (unsigned short)(u >> 16);
}

static __device__ __forceinline__ v8f wmma_bf16(v16bf a, v16bf b, v8f c) {
  return __builtin_amdgcn_wmma_f32_16x16x32_bf16(false, a, false, b, (short)0, c,
                                                 false, false);
}

// A-fragment (16x32 bf16) from row-major buffer, ld in halves; base -> (row0,k0).
// ISA layout: lane L holds row m=L&15, K in {kg..kg+7} and {16+kg..16+kg+7}, kg=(L>>4)*8.
static __device__ __forceinline__ v16bf load_afrag(const unsigned short* base, int ld) {
  const int lane = threadIdx.x & 31;
  const int m = lane & 15;
  const int kg = (lane >> 4) * 8;
  FragU f;
  f.q[0] = *(const uint4*)(base + m * ld + kg);
  f.q[1] = *(const uint4*)(base + m * ld + 16 + kg);
  return f.v;
}

// B-fragment (32x16 bf16) from B^T stored row-major (rows=N, contiguous K); base -> (n0,k0).
// ISA layout: lane L holds col n=L&15, K in {kg..kg+7} and {16+kg..16+kg+7}, kg=(L>>4)*8.
static __device__ __forceinline__ v16bf load_bfrag(const unsigned short* base, int ld) {
  const int lane = threadIdx.x & 31;
  const int n = lane & 15;
  const int kb = (lane >> 4) * 8;
  const unsigned short* p = base + n * ld + kb;
  FragU f;
  f.q[0] = *(const uint4*)(p);
  f.q[1] = *(const uint4*)(p + 16);
  return f.v;
}

__global__ __launch_bounds__(256) void cvt_bf16x4(const float* __restrict__ in,
                                                  unsigned short* __restrict__ out,
                                                  int n) {
  const int i = (blockIdx.x * 256 + threadIdx.x) * 4;
  if (i < n) {
    const float4 v = *(const float4*)(in + i);
    ushort4 o;
    o.x = f2bf(v.x);
    o.y = f2bf(v.y);
    o.z = f2bf(v.z);
    o.w = f2bf(v.w);
    *(ushort4*)(out + i) = o;
  }
}

#define LDT 40  // LDS row stride in halves (80B, 16B-aligned, conflict padding)

// C[M][O] = A[M][K] * Bt[O][K]^T + bias
// Block tile 128x128 (8 waves as 4x2), wave tile 32x64 (2x4 WMMA), K-step 32,
// double-buffered LDS -> one barrier per K-step.
__global__ __launch_bounds__(256) void gemm_bf16_bias(
    const unsigned short* __restrict__ A, const unsigned short* __restrict__ Bt,
    const float* __restrict__ bias, float* __restrict__ C, int K, int O) {
  __shared__ unsigned short As[2][128 * LDT];
  __shared__ unsigned short Bs[2][128 * LDT];
  const int t = threadIdx.x;
  const int wave = t >> 5;
  const int wr = wave >> 1;  // 0..3 -> 32-row group
  const int wc = wave & 1;   // 0..1 -> 64-col group
  const int row0 = blockIdx.y * 128;
  const int col0 = blockIdx.x * 128;

  v8f acc[8] = {};

  // tile loaders: 128 rows x 32 halves each for A and B; 32B per thread per tile
  const int ra = t >> 1, ca = (t & 1) * 16;

  const unsigned short* gA = A + (size_t)(row0 + ra) * K + ca;
  const unsigned short* gB = Bt + (size_t)(col0 + ra) * K + ca;
  unsigned short* lA = As[0] + ra * LDT + ca;
  unsigned short* lB = Bs[0] + ra * LDT + ca;
  const int bufStride = 128 * LDT;

  {  // preload k-tile 0 into buffer 0
    const uint4* pa = (const uint4*)(gA);
    const uint4* pb = (const uint4*)(gB);
    uint4* da = (uint4*)(lA);
    uint4* db = (uint4*)(lB);
    da[0] = pa[0];
    da[1] = pa[1];
    db[0] = pb[0];
    db[1] = pb[1];
  }
  __syncthreads();

  int cur = 0;
  for (int kk = 0; kk < K; kk += 32) {
    if (kk + 32 < K) {  // stage next tile into the other buffer
      const uint4* pa = (const uint4*)(gA + kk + 32);
      const uint4* pb = (const uint4*)(gB + kk + 32);
      uint4* da = (uint4*)(lA + (cur ^ 1) * bufStride);
      uint4* db = (uint4*)(lB + (cur ^ 1) * bufStride);
      da[0] = pa[0];
      da[1] = pa[1];
      db[0] = pb[0];
      db[1] = pb[1];
    }
    if (kk + 64 < K) {  // L2 prefetch two tiles ahead
      __builtin_prefetch(gA + kk + 64, 0, 1);
      __builtin_prefetch(gB + kk + 64, 0, 1);
    }

    const unsigned short* as = As[cur];
    const unsigned short* bs = Bs[cur];
    v16bf a0 = load_afrag(as + (wr * 32 + 0) * LDT, LDT);
    v16bf a1 = load_afrag(as + (wr * 32 + 16) * LDT, LDT);
    v16bf b0 = load_bfrag(bs + (wc * 64 + 0) * LDT, LDT);
    v16bf b1 = load_bfrag(bs + (wc * 64 + 16) * LDT, LDT);
    v16bf b2 = load_bfrag(bs + (wc * 64 + 32) * LDT, LDT);
    v16bf b3 = load_bfrag(bs + (wc * 64 + 48) * LDT, LDT);
    acc[0] = wmma_bf16(a0, b0, acc[0]);
    acc[1] = wmma_bf16(a0, b1, acc[1]);
    acc[2] = wmma_bf16(a0, b2, acc[2]);
    acc[3] = wmma_bf16(a0, b3, acc[3]);
    acc[4] = wmma_bf16(a1, b0, acc[4]);
    acc[5] = wmma_bf16(a1, b1, acc[5]);
    acc[6] = wmma_bf16(a1, b2, acc[6]);
    acc[7] = wmma_bf16(a1, b3, acc[7]);
    __syncthreads();
    cur ^= 1;
  }

  const int lane = t & 31;
  const int hf = lane >> 4;
  const int nn = lane & 15;
#pragma unroll
  for (int i = 0; i < 2; ++i)
#pragma unroll
    for (int j = 0; j < 4; ++j) {
#pragma unroll
      for (int r = 0; r < 8; ++r) {
        const int row = row0 + wr * 32 + i * 16 + hf * 8 + r;
        const int col = col0 + wc * 64 + j * 16 + nn;
        C[(size_t)row * O + col] = acc[i * 4 + j][r] + bias[col];
      }
    }
}

// RMSNorm q,k (scale folded into q), scatter to Q[bh][n][64], K[bh][s][64], Vt[bh][64][s]
__global__ __launch_bounds__(32) void post_qkv(
    const float* __restrict__ qkv, const float* __restrict__ qn,
    const float* __restrict__ kn, unsigned short* __restrict__ Q,
    unsigned short* __restrict__ Kb, unsigned short* __restrict__ Vt) {
  const int gid = blockIdx.x;  // tok*16 + h
  const int h = gid & 15;
  const int tok = gid >> 4;
  const int b = tok >> 11;
  const int n = tok & 2047;
  const int lane = threadIdx.x;
  const float* base = qkv + (size_t)tok * 3072 + h * 64;
  const float q0 = base[lane], q1 = base[lane + 32];
  const float k0 = base[1024 + lane], k1 = base[1024 + lane + 32];
  const float v0 = base[2048 + lane], v1 = base[2048 + lane + 32];
  float sq = q0 * q0 + q1 * q1;
  float sk = k0 * k0 + k1 * k1;
#pragma unroll
  for (int m = 16; m >= 1; m >>= 1) {
    sq += __shfl_xor(sq, m);
    sk += __shfl_xor(sk, m);
  }
  const float rq = rsqrtf(sq * (1.f / 64.f) + 1e-6f) * 0.125f;  // * hd^-0.5
  const float rk = rsqrtf(sk * (1.f / 64.f) + 1e-6f);
  const int bh = b * 16 + h;
  unsigned short* qo = Q + ((size_t)bh * 2048 + n) * 64;
  qo[lane] = f2bf(q0 * rq * qn[lane]);
  qo[lane + 32] = f2bf(q1 * rq * qn[lane + 32]);
  unsigned short* ko = Kb + ((size_t)bh * 2304 + n) * 64;
  ko[lane] = f2bf(k0 * rk * kn[lane]);
  ko[lane + 32] = f2bf(k1 * rk * kn[lane + 32]);
  unsigned short* vo = Vt + (size_t)bh * 64 * 2304 + n;
  vo[(size_t)lane * 2304] = f2bf(v0);
  vo[(size_t)(lane + 32) * 2304] = f2bf(v1);
}

__global__ __launch_bounds__(32) void post_kv(
    const float* __restrict__ kv, const float* __restrict__ kn,
    unsigned short* __restrict__ Kb, unsigned short* __restrict__ Vt) {
  const int gid = blockIdx.x;
  const int h = gid & 15;
  const int tok = gid >> 4;
  const int b = tok >> 8;
  const int m = tok & 255;
  const int s = 2048 + m;
  const int lane = threadIdx.x;
  const float* base = kv + (size_t)tok * 2048 + h * 64;
  const float k0 = base[lane], k1 = base[lane + 32];
  const float v0 = base[1024 + lane], v1 = base[1024 + lane + 32];
  float sk = k0 * k0 + k1 * k1;
#pragma unroll
  for (int mm = 16; mm >= 1; mm >>= 1) sk += __shfl_xor(sk, mm);
  const float rk = rsqrtf(sk * (1.f / 64.f) + 1e-6f);
  const int bh = b * 16 + h;
  unsigned short* ko = Kb + ((size_t)bh * 2304 + s) * 64;
  ko[lane] = f2bf(k0 * rk * kn[lane]);
  ko[lane + 32] = f2bf(k1 * rk * kn[lane + 32]);
  unsigned short* vo = Vt + (size_t)bh * 64 * 2304 + s;
  vo[(size_t)lane * 2304] = f2bf(v0);
  vo[(size_t)(lane + 32) * 2304] = f2bf(v1);
}

// Flash attention: 1 block = (bh, 64 q rows); each wave = 16 q rows; S chunks of 64
__global__ __launch_bounds__(128) void attn_fa(
    const unsigned short* __restrict__ Q, const unsigned short* __restrict__ Kb,
    const unsigned short* __restrict__ Vt, unsigned short* __restrict__ Aout) {
  __shared__ unsigned short Pl[4][16 * 72];
  const int t = threadIdx.x;
  const int wave = t >> 5;
  const int lane = t & 31;
  const int hf = lane >> 4;
  const int nl = lane & 15;

  const int bid = blockIdx.x;
  const int tile = bid & 31;  // N/64 tiles
  const int bh = bid >> 5;    // 0..31
  const int n0 = tile * 64 + wave * 16;

  const unsigned short* qbase = Q + ((size_t)bh * 2048 + n0) * 64;
  v16bf qf0 = load_afrag(qbase, 64);       // d = 0..31
  v16bf qf1 = load_afrag(qbase + 32, 64);  // d = 32..63

  v8f accO[4] = {};
  float mrow[8], lrow[8];
#pragma unroll
  for (int r = 0; r < 8; ++r) {
    mrow[r] = -1e30f;
    lrow[r] = 0.f;
  }

  unsigned short* P = Pl[wave];
  const unsigned short* kbh = Kb + (size_t)bh * 2304 * 64;
  const unsigned short* vbh = Vt + (size_t)bh * 64 * 2304;

  for (int s0 = 0; s0 < 2304; s0 += 64) {
    if (s0 + 64 < 2304) {  // L2 prefetch next K/V chunk
      __builtin_prefetch(kbh + (size_t)(s0 + 64) * 64, 0, 1);
      __builtin_prefetch(vbh + s0 + 64, 0, 1);
    }
    v8f lg[4] = {};
#pragma unroll
    for (int sub = 0; sub < 4; ++sub) {
      const unsigned short* kb = kbh + (size_t)(s0 + sub * 16) * 64;
      lg[sub] = wmma_bf16(qf0, load_bfrag(kb, 64), lg[sub]);
      lg[sub] = wmma_bf16(qf1, load_bfrag(kb + 32, 64), lg[sub]);
    }
    // online softmax (row = hf*8 + r, cols spread over 16 lanes of the half)
#pragma unroll
    for (int r = 0; r < 8; ++r) {
      float vmax = fmaxf(fmaxf(lg[0][r], lg[1][r]), fmaxf(lg[2][r], lg[3][r]));
#pragma unroll
      for (int msk = 8; msk >= 1; msk >>= 1) vmax = fmaxf(vmax, __shfl_xor(vmax, msk));
      const float mn = fmaxf(mrow[r], vmax);
      const float alpha = __expf(mrow[r] - mn);
      mrow[r] = mn;
      float rs = 0.f;
#pragma unroll
      for (int sub = 0; sub < 4; ++sub) {
        const float p = __expf(lg[sub][r] - mn);
        lg[sub][r] = p;
        rs += p;
      }
#pragma unroll
      for (int msk = 8; msk >= 1; msk >>= 1) rs += __shfl_xor(rs, msk);
      lrow[r] = lrow[r] * alpha + rs;
#pragma unroll
      for (int sub = 0; sub < 4; ++sub) accO[sub][r] *= alpha;
    }
    // C-layout -> LDS -> A-layout for P (wave-private tile; LDS ops of a wave
    // complete in order, one barrier is enough for cross-lane visibility)
#pragma unroll
    for (int sub = 0; sub < 4; ++sub)
#pragma unroll
      for (int r = 0; r < 8; ++r)
        P[(hf * 8 + r) * 72 + sub * 16 + nl] = f2bf(lg[sub][r]);
    __syncthreads();
    v16bf p0 = load_afrag(P, 72);       // k(s) = 0..31
    v16bf p1 = load_afrag(P + 32, 72);  // k(s) = 32..63
#pragma unroll
    for (int sub = 0; sub < 4; ++sub) {
      const unsigned short* vb = vbh + (size_t)(sub * 16) * 2304 + s0;
      accO[sub] = wmma_bf16(p0, load_bfrag(vb, 2304), accO[sub]);
      accO[sub] = wmma_bf16(p1, load_bfrag(vb + 32, 2304), accO[sub]);
    }
  }

  const int b = bh >> 4;
  const int h = bh & 15;
#pragma unroll
  for (int sub = 0; sub < 4; ++sub) {
#pragma unroll
    for (int r = 0; r < 8; ++r) {
      const int tok = (b << 11) + n0 + hf * 8 + r;
      const int col = h * 64 + sub * 16 + nl;
      Aout[(size_t)tok * 1024 + col] = f2bf(accO[sub][r] / lrow[r]);
    }
  }
}

extern "C" void kernel_launch(void* const* d_in, const int* in_sizes, int n_in,
                              void* d_out, int out_size, void* d_ws, size_t ws_size,
                              hipStream_t stream) {
  (void)in_sizes; (void)n_in; (void)out_size; (void)ws_size;
  const float* x = (const float*)d_in[0];
  const float* y = (const float*)d_in[1];
  const float* qkv_w = (const float*)d_in[2];
  const float* qkv_b = (const float*)d_in[3];
  const float* kv_w = (const float*)d_in[4];
  const float* kv_b = (const float*)d_in[5];
  const float* qn_w = (const float*)d_in[6];
  const float* kn_w = (const float*)d_in[7];
  const float* proj_w = (const float*)d_in[8];
  const float* proj_b = (const float*)d_in[9];
  float* out = (float*)d_out;

  char* ws = (char*)d_ws;
  size_t off = 0;
  auto take = [&](size_t bytes) {
    void* p = ws + off;
    off += (bytes + 255) & ~(size_t)255;
    return p;
  };

  unsigned short* xb = (unsigned short*)take((size_t)4096 * 1024 * 2);
  unsigned short* yb = (unsigned short*)take((size_t)512 * 1024 * 2);
  unsigned short* wqkv = (unsigned short*)take((size_t)3072 * 1024 * 2);
  unsigned short* wkv = (unsigned short*)take((size_t)2048 * 1024 * 2);
  unsigned short* wprj = (unsigned short*)take((size_t)1024 * 1024 * 2);
  float* qkvraw = (float*)take((size_t)4096 * 3072 * 4);
  float* kvraw = (float*)take((size_t)512 * 2048 * 4);
  unsigned short* Qb = (unsigned short*)take((size_t)32 * 2048 * 64 * 2);
  unsigned short* Kbf = (unsigned short*)take((size_t)32 * 2304 * 64 * 2);
  unsigned short* Vtb = (unsigned short*)take((size_t)32 * 2304 * 64 * 2);
  unsigned short* att = (unsigned short*)take((size_t)4096 * 1024 * 2);

  cvt_bf16x4<<<(4096 * 1024) / 1024, 256, 0, stream>>>(x, xb, 4096 * 1024);
  cvt_bf16x4<<<(512 * 1024) / 1024, 256, 0, stream>>>(y, yb, 512 * 1024);
  cvt_bf16x4<<<(3072 * 1024) / 1024, 256, 0, stream>>>(qkv_w, wqkv, 3072 * 1024);
  cvt_bf16x4<<<(2048 * 1024) / 1024, 256, 0, stream>>>(kv_w, wkv, 2048 * 1024);
  cvt_bf16x4<<<(1024 * 1024) / 1024, 256, 0, stream>>>(proj_w, wprj, 1024 * 1024);

  gemm_bf16_bias<<<dim3(3072 / 128, 4096 / 128), 256, 0, stream>>>(xb, wqkv, qkv_b,
                                                                   qkvraw, 1024, 3072);
  gemm_bf16_bias<<<dim3(2048 / 128, 512 / 128), 256, 0, stream>>>(yb, wkv, kv_b, kvraw,
                                                                  1024, 2048);

  post_qkv<<<4096 * 16, 32, 0, stream>>>(qkvraw, qn_w, kn_w, Qb, Kbf, Vtb);
  post_kv<<<512 * 16, 32, 0, stream>>>(kvraw, kn_w, Kbf, Vtb);

  attn_fa<<<32 * 32, 128, 0, stream>>>(Qb, Kbf, Vtb, att);

  gemm_bf16_bias<<<dim3(1024 / 128, 4096 / 128), 256, 0, stream>>>(att, wprj, proj_b,
                                                                   out, 1024, 1024);
}
